// allocationNN_LSTM_11390253269777
// MI455X (gfx1250) — matrix operand order repaired and checked by reading, not compiled, not run on hardware
//
#include <hip/hip_runtime.h>
#include <hip/hip_bf16.h>

typedef __attribute__((ext_vector_type(16))) _Float16 v16h;
typedef __attribute__((ext_vector_type(8)))  _Float16 v8h;
typedef __attribute__((ext_vector_type(8)))  float    v8f;

#define LSTM_H 64
#define LSTM_T 1024
#define LSTM_B 1024

#define LOG2E_F 1.44269504088896340736f

// Branchless hardware transcendentals: v_exp_f32 / v_rcp_f32 / v_tanh_f32.
__device__ __forceinline__ float fast_exp(float x) {
  return __builtin_amdgcn_exp2f(x * LOG2E_F);
}
__device__ __forceinline__ float sigm(float x) {
  return __builtin_amdgcn_rcpf(1.0f + fast_exp(-x));
}
__device__ __forceinline__ float fast_tanh(float x) {
#if __has_builtin(__builtin_amdgcn_tanh_f32)
  return __builtin_amdgcn_tanh_f32(x);
#else
  // tanh(x) = 1 - 2/(e^{2x}+1); exp2 overflow -> inf -> rcp -> 0 -> +1, underflow -> -1.
  return 1.0f - 2.0f * __builtin_amdgcn_rcpf(fast_exp(2.0f * x) + 1.0f);
#endif
}

// B fragment for v_wmma_f32_16x16x32_f16: B = W^T, W row-major [256][64].
// lane holds column col; element e -> K = kc*32 + e + 16*(lane>=16)  (contiguous run)
__device__ __forceinline__ v16h load_bfrag(const float* __restrict__ W, int col, int kc, int lane) {
  const float* p = W + col * 64 + kc * 32 + ((lane >> 4) << 4);
  v16h f;
#pragma unroll
  for (int e = 0; e < 16; ++e) f[e] = (_Float16)p[e];
  return f;
}

// A fragment (16x32 f16) from LDS h[16][64]:
// lane row m = lane&15; K = kc*32 + e + 16*(lane>=16)
// => one contiguous 16-half run per lane, loaded as two 8-half (16B) loads.
__device__ __forceinline__ v16h load_afrag(const _Float16* __restrict__ h, int kc, int lane) {
  int m  = lane & 15;
  int kb = kc * 32 + ((lane >> 4) << 4);
  v8h lo = *(const v8h*)(h + m * 64 + kb);
  v8h hi = *(const v8h*)(h + m * 64 + kb + 8);
  v16h a;
#pragma unroll
  for (int e = 0; e < 8; ++e) { a[e] = lo[e]; a[e + 8] = hi[e]; }
  return a;
}

#define WMMA_F16(a, b, c) \
  __builtin_amdgcn_wmma_f32_16x16x32_f16(false, (a), false, (b), (short)0, (c), false, false)

__global__ __launch_bounds__(256, 1)
void lstm_autoencoder_kernel(
    const float* __restrict__ x,
    const float* __restrict__ e1_Wih, const float* __restrict__ e1_Whh,
    const float* __restrict__ e1_bih, const float* __restrict__ e1_bhh,
    const float* __restrict__ e2_Wih, const float* __restrict__ e2_Whh,
    const float* __restrict__ e2_bih, const float* __restrict__ e2_bhh,
    const float* __restrict__ enc_fc_W, const float* __restrict__ enc_fc_b,
    const float* __restrict__ d1_Wih, const float* __restrict__ d1_bih, const float* __restrict__ d1_bhh,
    const float* __restrict__ d2_Wih, const float* __restrict__ d2_bih, const float* __restrict__ d2_bhh,
    const float* __restrict__ dec_fc_W, const float* __restrict__ dec_fc_b,
    float* __restrict__ out)
{
  __shared__ __align__(16) _Float16 h1buf[2][16 * 64]; // double-buffered layer1 h (f16)
  __shared__ __align__(16) _Float16 h2buf[16 * 64];    // layer2 h (f16)
  __shared__ __align__(16) float    h2f32[16 * 64];    // final encoder h2 (f32), reused by decoder
  __shared__ __align__(16) float    h1dec[16 * 64];    // decoder layer1 h
  __shared__            float       uls[16 * 8];       // bottleneck u (padded 5->8)

  const int tid   = threadIdx.x;
  const int lane  = tid & 31;
  const int wave  = tid >> 5;
  const int b0    = blockIdx.x * 16;
  const int n     = lane & 15;          // WMMA column owned by this lane
  const int mbase = (lane >> 4) << 3;   // row base for C-layout (rows mbase..mbase+7)

  const bool isL1 = (wave < 4);         // waves 0-3: encoder layer1; 4-7: encoder layer2
  const int  wl   = wave & 3;           // hidden slice [wl*16, wl*16+16)

  // ---- zero-init LDS h state ----
  {
    _Float16* p = &h1buf[0][0];
    for (int i = tid; i < 2 * 16 * 64; i += 256) p[i] = (_Float16)0.f;
    for (int i = tid; i < 16 * 64; i += 256) h2buf[i] = (_Float16)0.f;
  }

  // ---- load loop-invariant weight fragments into registers ----
  // bw[q][0..1]: L1 -> Whh1 (k-chunks 0,1) ; L2 -> Wih2 (k-chunks 0,1)
  // bw[q][2..3]: L2 -> Whh2 (k-chunks 0,1) ; unused by L1
  v16h  bw[4][4];
  float wih1[4][3];
  float bias[4];
  float cst[8];
#pragma unroll
  for (int r = 0; r < 8; ++r) cst[r] = 0.f;

#pragma unroll
  for (int q = 0; q < 4; ++q) {
    const int col = q * 64 + wl * 16 + n;
    if (isL1) {
      bw[q][0] = load_bfrag(e1_Whh, col, 0, lane);
      bw[q][1] = load_bfrag(e1_Whh, col, 1, lane);
      wih1[q][0] = e1_Wih[col * 3 + 0];
      wih1[q][1] = e1_Wih[col * 3 + 1];
      wih1[q][2] = e1_Wih[col * 3 + 2];
      bias[q] = e1_bih[col] + e1_bhh[col];
    } else {
      bw[q][0] = load_bfrag(e2_Wih, col, 0, lane);
      bw[q][1] = load_bfrag(e2_Wih, col, 1, lane);
      bw[q][2] = load_bfrag(e2_Whh, col, 0, lane);
      bw[q][3] = load_bfrag(e2_Whh, col, 1, lane);
      bias[q] = e2_bih[col] + e2_bhh[col];
    }
  }

  // ---- pipelined recurrence: layer1 runs step t, layer2 runs step t-1 ----
  for (int t = 0; t <= LSTM_T; ++t) {
    __syncthreads(); // separates previous iteration's writes from this iteration's reads

    v16h aA0 = {}, aA1 = {}, aB0 = {}, aB1 = {};
    float xv[8][3];
    bool active;

    if (isL1) {
      active = (t < LSTM_T);
      if (active) {
        const _Float16* hprev = &h1buf[(t + 1) & 1][0]; // h1 state (t-1)
        aA0 = load_afrag(hprev, 0, lane);
        aA1 = load_afrag(hprev, 1, lane);
#pragma unroll
        for (int r = 0; r < 8; ++r) {
          const int row = b0 + mbase + r;
#pragma unroll
          for (int i = 0; i < 3; ++i)
            xv[r][i] = x[row * (LSTM_T * 3) + t * 3 + i];
        }
      }
    } else {
      active = (t >= 1);
      if (active) {
        const _Float16* h1cur = &h1buf[(t + 1) & 1][0]; // h1 state (t-1) written last iter
        aA0 = load_afrag(h1cur, 0, lane);
        aA1 = load_afrag(h1cur, 1, lane);
        aB0 = load_afrag(h2buf, 0, lane);               // h2 state (t-2)
        aB1 = load_afrag(h2buf, 1, lane);
      }
    }

    __syncthreads(); // all fragment reads done before anyone rewrites state

    if (isL1) {
      if (active) {
        v8f acc[4];
#pragma unroll
        for (int q = 0; q < 4; ++q) {
          v8f a;
#pragma unroll
          for (int r = 0; r < 8; ++r)
            a[r] = bias[q] + xv[r][0] * wih1[q][0] + xv[r][1] * wih1[q][1] + xv[r][2] * wih1[q][2];
          a = WMMA_F16(aA0, bw[q][0], a);
          a = WMMA_F16(aA1, bw[q][1], a);
          acc[q] = a;
        }
        _Float16* hdst = &h1buf[t & 1][0];
#pragma unroll
        for (int r = 0; r < 8; ++r) {
          float iv = sigm(acc[0][r]);
          float fv = sigm(acc[1][r]);
          float gv = fast_tanh(acc[2][r]);
          float ov = sigm(acc[3][r]);
          cst[r] = fv * cst[r] + iv * gv;
          float hv = ov * fast_tanh(cst[r]);
          hdst[(mbase + r) * 64 + wl * 16 + n] = (_Float16)hv;
        }
      }
    } else {
      if (active) {
        v8f acc[4];
#pragma unroll
        for (int q = 0; q < 4; ++q) {
          v8f a;
#pragma unroll
          for (int r = 0; r < 8; ++r) a[r] = bias[q];
          a = WMMA_F16(aA0, bw[q][0], a);  // h1_t  @ Wih2^T
          a = WMMA_F16(aA1, bw[q][1], a);
          a = WMMA_F16(aB0, bw[q][2], a);  // h2_{t-1} @ Whh2^T
          a = WMMA_F16(aB1, bw[q][3], a);
          acc[q] = a;
        }
#pragma unroll
        for (int r = 0; r < 8; ++r) {
          float iv = sigm(acc[0][r]);
          float fv = sigm(acc[1][r]);
          float gv = fast_tanh(acc[2][r]);
          float ov = sigm(acc[3][r]);
          cst[r] = fv * cst[r] + iv * gv;
          float hv = ov * fast_tanh(cst[r]);
          const int off = (mbase + r) * 64 + wl * 16 + n;
          h2buf[off] = (_Float16)hv;
          if (t == LSTM_T) h2f32[off] = hv; // final encoder state, fp32
        }
      }
    }
  }

  __syncthreads();

  // ---- bottleneck: u = h2_last @ enc_fc_W^T + enc_fc_b  (enc_fc_W: [5][64]) ----
  if (tid < 80) {
    const int m = tid / 5, j = tid % 5;
    float s = enc_fc_b[j];
#pragma unroll 8
    for (int k = 0; k < 64; ++k) s += h2f32[m * 64 + k] * enc_fc_W[j * 64 + k];
    uls[m * 8 + j] = s;
    out[(b0 + m) * 5 + j] = s;  // output u (B,1,5)
  }
  __syncthreads();

  // ---- decoder layer1 (T=1, h0=c0=0 => no Whh term) ----
  for (int idx = tid; idx < 16 * 64; idx += 256) {
    const int m = idx >> 6, k = idx & 63;
    float g[4];
#pragma unroll
    for (int q = 0; q < 4; ++q) {
      const int col = q * 64 + k;
      float s = d1_bih[col] + d1_bhh[col];
#pragma unroll
      for (int i = 0; i < 5; ++i) s += uls[m * 8 + i] * d1_Wih[col * 5 + i];
      g[q] = s;
    }
    const float cc = sigm(g[0]) * fast_tanh(g[2]);
    h1dec[m * 64 + k] = sigm(g[3]) * fast_tanh(cc);
  }
  __syncthreads();

  // ---- decoder layer2 (T=1, h0=c0=0) ----
  for (int idx = tid; idx < 16 * 64; idx += 256) {
    const int m = idx >> 6, k = idx & 63;
    float g[4];
#pragma unroll
    for (int q = 0; q < 4; ++q) {
      const int col = q * 64 + k;
      float s = d2_bih[col] + d2_bhh[col];
#pragma unroll 8
      for (int kk = 0; kk < 64; ++kk) s += h1dec[m * 64 + kk] * d2_Wih[col * 64 + kk];
      g[q] = s;
    }
    const float cc = sigm(g[0]) * fast_tanh(g[2]);
    h2f32[m * 64 + k] = sigm(g[3]) * fast_tanh(cc); // reuse buffer for decoder h2
  }
  __syncthreads();

  // ---- head: tau = h2d @ dec_fc_W^T + dec_fc_b  (dec_fc_W: [3][64]) ----
  if (tid < 48) {
    const int m = tid / 3, j = tid % 3;
    float s = dec_fc_b[j];
#pragma unroll 8
    for (int k = 0; k < 64; ++k) s += h2f32[m * 64 + k] * dec_fc_W[j * 64 + k];
    out[LSTM_B * 5 + (b0 + m) * 3 + j] = s;  // output tau (B,3) after u
  }
}

extern "C" void kernel_launch(void* const* d_in, const int* in_sizes, int n_in,
                              void* d_out, int out_size, void* d_ws, size_t ws_size,
                              hipStream_t stream) {
  const float* x        = (const float*)d_in[0];
  const float* e1_Wih   = (const float*)d_in[1];
  const float* e1_Whh   = (const float*)d_in[2];
  const float* e1_bih   = (const float*)d_in[3];
  const float* e1_bhh   = (const float*)d_in[4];
  const float* e2_Wih   = (const float*)d_in[5];
  const float* e2_Whh   = (const float*)d_in[6];
  const float* e2_bih   = (const float*)d_in[7];
  const float* e2_bhh   = (const float*)d_in[8];
  const float* enc_fc_W = (const float*)d_in[9];
  const float* enc_fc_b = (const float*)d_in[10];
  const float* d1_Wih   = (const float*)d_in[11];
  // d_in[12] = d1_Whh (unused: decoder h0 = 0, T = 1)
  const float* d1_bih   = (const float*)d_in[13];
  const float* d1_bhh   = (const float*)d_in[14];
  const float* d2_Wih   = (const float*)d_in[15];
  // d_in[16] = d2_Whh (unused)
  const float* d2_bih   = (const float*)d_in[17];
  const float* d2_bhh   = (const float*)d_in[18];
  const float* dec_fc_W = (const float*)d_in[19];
  const float* dec_fc_b = (const float*)d_in[20];
  float* outp = (float*)d_out;

  dim3 grid(LSTM_B / 16);   // 64 batch tiles
  dim3 block(256);          // 8 waves: 4 for layer1, 4 for layer2 (pipelined)
  lstm_autoencoder_kernel<<<grid, block, 0, stream>>>(
      x, e1_Wih, e1_Whh, e1_bih, e1_bhh,
      e2_Wih, e2_Whh, e2_bih, e2_bhh,
      enc_fc_W, enc_fc_b,
      d1_Wih, d1_bih, d1_bhh,
      d2_Wih, d2_bih, d2_bhh,
      dec_fc_W, dec_fc_b, outp);
}
